// AttentionBasedPolicyNetwork_20572893348171
// MI455X (gfx1250) — compile-verified, matrix operand order checked by reading, not run on hardware
//
#include <hip/hip_runtime.h>
#include <math.h>

// ---------------------------------------------------------------------------
// AttentionBasedPolicyNetwork for MI455X (gfx1250, wave32, WMMA 16x16x32 f16)
//
//  * Generic strided-batched WMMA GEMM (f16 A/B, f32 accum, bias/alpha/
//    relu/tanh epilogue, f16 or f32 out) covers all Linear layers, QK^T and
//    P*V. LDS staging uses 128-bit copies; gfx1250 async-to-LDS builtin
//    (ASYNCcnt + s_wait_asynccnt) used when available.
//  * Fused compat kernel: h1 = relu(a_t + b_r + c1_b) tile in LDS (f16),
//    h2 GEMM via v_wmma from LDS, c3 dot + sigmoid in-register. h1/h2 never
//    touch HBM (saves ~270MB of traffic vs. materializing [B,T,R,D]).
//  * Embedding+LayerNorm fused; per-row softmax; tiny-N heads scalar.
//  * Attention streams write straight into the [2048,1024] concat buffer
//    via ldc=1024 column offsets.
// ---------------------------------------------------------------------------

typedef __attribute__((ext_vector_type(16))) _Float16 v16h;
typedef __attribute__((ext_vector_type(8)))  float    v8f;
typedef int v4i __attribute__((vector_size(16)));
typedef _Float16 half_t;

#define B_ 4
#define T_ 512
#define R_ 128
#define D_ 256

#define BM 64
#define BN 64
#define BK 32
#define LDSK 40  // padded LDS row stride in halves (80B, 16B-aligned rows)

// ---- 128-bit global->LDS staging (async if available) ----------------------
#if defined(__has_builtin)
#if __has_builtin(__builtin_amdgcn_global_load_async_to_lds_b128)
#define USE_ASYNC_LDS 1
#endif
#endif

__device__ __forceinline__ void stage16(const half_t* gp, half_t* lp) {
#ifdef USE_ASYNC_LDS
  // Builtin signature (from clang diagnostics): param0 = AS1 v4i*, then LDS
  // dst, imm offset, imm cpol. Low 32 bits of a generic LDS pointer are the
  // LDS byte offset, so the inttoptr cast below carries the right address.
  __builtin_amdgcn_global_load_async_to_lds_b128(
      (__attribute__((address_space(1))) v4i*)(unsigned long long)gp,
      (__attribute__((address_space(3))) v4i*)(unsigned long long)lp,
      0, 0);
#else
  *reinterpret_cast<uint4*>(lp) = *reinterpret_cast<const uint4*>(gp);
#endif
}

__device__ __forceinline__ void async_fence() {
#ifdef USE_ASYNC_LDS
#if __has_builtin(__builtin_amdgcn_s_wait_asynccnt)
  __builtin_amdgcn_s_wait_asynccnt(0);
#else
  asm volatile("s_wait_asynccnt 0x0" ::: "memory");
#endif
#endif
}

// ---------------------------------------------------------------------------
// Generic strided batched GEMM:  C[z] = act(alpha * A[z] @ B[z]^T + bias)
// A: f16 [M,K] (lda). B: if BT: f16 [N,K] (ldb); else f16 [K,N] (ldb).
// Batch z -> (b,h) = (z/heads, z%heads); operand offset = b*s?B + h*s?H.
// ---------------------------------------------------------------------------
template <int ACT, int BT, int OF16>
__global__ void __launch_bounds__(256)
gemm_wmma(const half_t* __restrict__ A, int lda, long long saB, long long saH,
          const half_t* __restrict__ Bm, int ldb, long long sbB, long long sbH,
          float* __restrict__ Cf, half_t* __restrict__ Ch, int ldc,
          long long scB, long long scH,
          int M, int N, int K, int heads,
          const float* __restrict__ bias, float alpha) {
  __shared__ alignas(16) half_t As[BM][LDSK];
  __shared__ alignas(16) half_t Bs[BN][LDSK];

  const int z  = blockIdx.z;
  const int bb = z / heads, hh = z % heads;
  A  += (size_t)bb * saB + (size_t)hh * saH;
  Bm += (size_t)bb * sbB + (size_t)hh * sbH;
  const size_t coff = (size_t)bb * scB + (size_t)hh * scH;

  const int m_blk = blockIdx.x * BM;
  const int n_blk = blockIdx.y * BN;
  const int tid   = threadIdx.x;
  const int lane  = tid & 31;
  const int wave  = tid >> 5;
  const int wm    = (wave & 3) * 16;   // wave's m tile inside block
  const int wn    = (wave >> 2) * 32;  // wave's 2 n tiles inside block

  const bool fastA = (((unsigned long long)A & 15ull) == 0) && ((lda & 7) == 0) &&
                     (m_blk + BM <= M);
  const bool fastB = BT && (((unsigned long long)Bm & 15ull) == 0) &&
                     ((ldb & 7) == 0) && (n_blk + BN <= N);
  // fast-path per-thread staging coords: 64 rows x 4 chunks of 8 halves
  const int srow = tid >> 2;
  const int scol = (tid & 3) * 8;

  v8f acc0 = {};
  v8f acc1 = {};

  for (int k0 = 0; k0 < K; k0 += BK) {
    const bool kfull = (k0 + BK <= K);
    __syncthreads();
    // ---- stage A tile (64x32) ----
    if (fastA && kfull) {
      const half_t* gp = A + (size_t)(m_blk + srow) * lda + k0 + scol;
      stage16(gp, &As[srow][scol]);
      if (k0 + BK < K) __builtin_prefetch(gp + BK, 0, 1);
    } else {
      for (int i = 0; i < 8; ++i) {
        int idx = i * 256 + tid;
        int row = idx >> 5, col = idx & 31;
        int rg = m_blk + row, kg = k0 + col;
        half_t v = (half_t)0.f;
        if (rg < M && kg < K) v = A[(size_t)rg * lda + kg];
        As[row][col] = v;
      }
    }
    // ---- stage B tile (64x32 as [n][k]) ----
    if (fastB && kfull) {
      const half_t* gp = Bm + (size_t)(n_blk + srow) * ldb + k0 + scol;
      stage16(gp, &Bs[srow][scol]);
      if (k0 + BK < K) __builtin_prefetch(gp + BK, 0, 1);
    } else {
      for (int i = 0; i < 8; ++i) {
        int idx = i * 256 + tid;
        int row = idx >> 5, col = idx & 31;  // row = n, col = k
        int ng = n_blk + row, kg = k0 + col;
        half_t v = (half_t)0.f;
        if (ng < N && kg < K)
          v = BT ? Bm[(size_t)ng * ldb + kg] : Bm[(size_t)kg * ldb + ng];
        Bs[row][col] = v;
      }
    }
    async_fence();
    __syncthreads();

    // ---- fragments (ISA 16-bit 16x32 layout) + WMMA ----
    const int kob = (lane >> 4) * 8;
    const int mf  = wm + (lane & 15);
    const int nf  = wn + (lane & 15);
    v16h af, bf0, bf1;
#pragma unroll
    for (int e = 0; e < 16; ++e) {
      int g = e >> 1;
      int k = ((g & 3) * 2 + (e & 1)) + kob + ((g & 4) ? 16 : 0);
      af[e]  = As[mf][k];
      bf0[e] = Bs[nf][k];
      bf1[e] = Bs[nf + 16][k];
    }
    acc0 = __builtin_amdgcn_wmma_f32_16x16x32_f16(false, af, false, bf0,
                                                  (short)0, acc0, false, false);
    acc1 = __builtin_amdgcn_wmma_f32_16x16x32_f16(false, af, false, bf1,
                                                  (short)0, acc1, false, false);
  }

  // Epilogue. C layout: lane0-15 -> N=lane, M=r; lane16-31 -> N=lane-16, M=r+8.
  const int mbase = m_blk + wm + (lane >> 4) * 8;
  const int n0    = n_blk + wn + (lane & 15);
  const int n1    = n0 + 16;
#pragma unroll
  for (int r = 0; r < 8; ++r) {
    int m = mbase + r;
    if (m >= M) continue;
    for (int half = 0; half < 2; ++half) {
      int n   = half ? n1 : n0;
      float v = half ? acc1[r] : acc0[r];
      if (n >= N) continue;
      v *= alpha;
      if (bias) v += bias[n];
      if (ACT == 1) v = fmaxf(v, 0.f);
      if (ACT == 2) v = tanhf(v);
      size_t o = coff + (size_t)m * ldc + n;
      if (OF16) Ch[o] = (half_t)v;
      else      Cf[o] = v;
    }
  }
}

// ---------------------------------------------------------------------------
// Fused pairwise compatibility kernel.
// Tile: 16 tasks x 4 resources = 64 h1 rows of 256 dims in LDS (f16),
// h2 = relu(h1 @ c2^T + c2_b) via WMMA, compat = sigmoid(h2 . c3 + c3_b).
// ---------------------------------------------------------------------------
__global__ void __launch_bounds__(256)
compat_fused(const float* __restrict__ aC,   // [B,T,256] = te @ c1a^T
             const float* __restrict__ bC,   // [B,R,256] = re @ c1b^T
             const float* __restrict__ c1b,
             const half_t* __restrict__ c2w, // f16 [128,256]
             const float* __restrict__ c2b,
             const float* __restrict__ c3w,  // f32 [128]
             const float* __restrict__ c3b,
             float* __restrict__ out)        // [B,T,R]
{
  __shared__ alignas(16) half_t h1s[64][264];  // row stride 528B (16B mult)
  __shared__ alignas(16) half_t Bs[128][LDSK];
  __shared__ float pc3[64];

  const int b   = blockIdx.z;
  const int t0  = blockIdx.x * 16;
  const int r0  = blockIdx.y * 4;
  const int tid = threadIdx.x;

  // Build h1 tile: each thread fills 64 contiguous dims of one row.
  {
    int p  = tid >> 2;            // pair row (tl*4 + rl)
    int d0 = (tid & 3) * 64;
    int tl = p >> 2, rl = p & 3;
    const float* ap = &aC[((size_t)b * T_ + (t0 + tl)) * D_ + d0];
    const float* bp = &bC[((size_t)b * R_ + (r0 + rl)) * D_ + d0];
    const float* cb = &c1b[d0];
#pragma unroll 8
    for (int j = 0; j < 64; ++j) {
      float v = ap[j] + bp[j] + cb[j];
      h1s[p][d0 + j] = (half_t)fmaxf(v, 0.f);
    }
  }
  if (tid < 64) pc3[tid] = 0.f;

  const int lane = tid & 31;
  const int wave = tid >> 5;
  const int wm   = (wave & 3) * 16;   // 4 m tiles of 16
  const int wn   = (wave >> 2) * 64;  // 2 n groups of 64 (4 tiles each)
  v8f acc[4] = {v8f{}, v8f{}, v8f{}, v8f{}};

  const int crow = tid >> 1;          // c2 staging: 128 rows, 2 threads/row
  const int ccol = (tid & 1) * 16;

  for (int k0 = 0; k0 < 256; k0 += BK) {
    __syncthreads();
    // Stage c2 tile [128 n][32 k] with 128-bit copies.
    stage16(&c2w[(size_t)crow * 256 + k0 + ccol], &Bs[crow][ccol]);
    stage16(&c2w[(size_t)crow * 256 + k0 + ccol + 8], &Bs[crow][ccol + 8]);
    async_fence();
    __syncthreads();

    const int kob = (lane >> 4) * 8;
    const int mf  = wm + (lane & 15);
    v16h af;
#pragma unroll
    for (int e = 0; e < 16; ++e) {
      int g = e >> 1;
      int k = ((g & 3) * 2 + (e & 1)) + kob + ((g & 4) ? 16 : 0);
      af[e] = h1s[mf][k0 + k];
    }
#pragma unroll
    for (int nt = 0; nt < 4; ++nt) {
      const int nf = wn + nt * 16 + (lane & 15);
      v16h bf;
#pragma unroll
      for (int e = 0; e < 16; ++e) {
        int g = e >> 1;
        int k = ((g & 3) * 2 + (e & 1)) + kob + ((g & 4) ? 16 : 0);
        bf[e] = Bs[nf][k];
      }
      acc[nt] = __builtin_amdgcn_wmma_f32_16x16x32_f16(false, af, false, bf,
                                                       (short)0, acc[nt],
                                                       false, false);
    }
  }

  // relu(h2 + c2_b) * c3_w, reduced per h1-row through LDS atomics.
  const int mbase = wm + (lane >> 4) * 8;
  float s[8] = {0, 0, 0, 0, 0, 0, 0, 0};
#pragma unroll
  for (int nt = 0; nt < 4; ++nt) {
    int n    = wn + nt * 16 + (lane & 15);
    float w3 = c3w[n];
    float b2 = c2b[n];
#pragma unroll
    for (int r = 0; r < 8; ++r) s[r] += fmaxf(acc[nt][r] + b2, 0.f) * w3;
  }
#pragma unroll
  for (int r = 0; r < 8; ++r) atomicAdd(&pc3[mbase + r], s[r]);
  __syncthreads();

  if (tid < 64) {
    float v = pc3[tid] + c3b[0];
    v = 1.f / (1.f + __expf(-v));
    int tl = tid >> 2, rl = tid & 3;
    out[((size_t)b * T_ + (t0 + tl)) * R_ + (r0 + rl)] = v;
  }
}

// ---------------------------------------------------------------------------
// Embedding + LayerNorm: out = LN(relu(x @ W^T + b)), F<=16, D=256.
// ---------------------------------------------------------------------------
__global__ void __launch_bounds__(256)
embed_ln(const float* __restrict__ X, int F, const float* __restrict__ W,
         const float* __restrict__ bv, const float* __restrict__ g,
         const float* __restrict__ be, float* __restrict__ outF,
         half_t* __restrict__ outH) {
  __shared__ float red[256];
  const int row = blockIdx.x;
  const int d   = threadIdx.x;
  const float* x = X + (size_t)row * F;
  float acc = bv[d];
  for (int f = 0; f < F; ++f) acc += x[f] * W[(size_t)d * F + f];
  acc = fmaxf(acc, 0.f);

  red[d] = acc;
  __syncthreads();
  for (int st = 128; st > 0; st >>= 1) {
    if (d < st) red[d] += red[d + st];
    __syncthreads();
  }
  float mean = red[0] * (1.f / 256.f);
  __syncthreads();
  float dd = acc - mean;
  red[d] = dd * dd;
  __syncthreads();
  for (int st = 128; st > 0; st >>= 1) {
    if (d < st) red[d] += red[d + st];
    __syncthreads();
  }
  float var = red[0] * (1.f / 256.f);
  float y = dd * rsqrtf(var + 1e-5f) * g[d] + be[d];
  outF[(size_t)row * 256 + d] = y;
  outH[(size_t)row * 256 + d] = (half_t)y;
}

// Row softmax, writes f16 probabilities.
__global__ void __launch_bounds__(256)
softmax_rows(const float* __restrict__ S, half_t* __restrict__ P, int L) {
  __shared__ float red[256];
  const int row = blockIdx.x;
  const float* s = S + (size_t)row * L;
  half_t* p = P + (size_t)row * L;
  float mx = -1e30f;
  for (int i = threadIdx.x; i < L; i += 256) mx = fmaxf(mx, s[i]);
  red[threadIdx.x] = mx;
  __syncthreads();
  for (int st = 128; st > 0; st >>= 1) {
    if (threadIdx.x < st)
      red[threadIdx.x] = fmaxf(red[threadIdx.x], red[threadIdx.x + st]);
    __syncthreads();
  }
  mx = red[0];
  __syncthreads();
  float sum = 0.f;
  for (int i = threadIdx.x; i < L; i += 256) sum += __expf(s[i] - mx);
  red[threadIdx.x] = sum;
  __syncthreads();
  for (int st = 128; st > 0; st >>= 1) {
    if (threadIdx.x < st) red[threadIdx.x] += red[threadIdx.x + st];
    __syncthreads();
  }
  float inv = 1.f / red[0];
  for (int i = threadIdx.x; i < L; i += 256)
    p[i] = (half_t)(__expf(s[i] - mx) * inv);
}

// x16[b,t,:] = f16((te + temp_enc[t]) * exp(-deadline/3600))
__global__ void __launch_bounds__(256)
temporal_x(const float* __restrict__ te, const float* __restrict__ tenc,
           const float* __restrict__ dl, half_t* __restrict__ x16) {
  const int row = blockIdx.x;  // b*T + t
  const int d   = threadIdx.x;
  const int t   = row & (T_ - 1);
  float urg = __expf(-dl[row] * (1.f / 3600.f));
  x16[(size_t)row * D_ + d] =
      (half_t)((te[(size_t)row * D_ + d] + tenc[(size_t)t * D_ + d]) * urg);
}

// penh16 = f16(te + pri_emb[prio])
__global__ void __launch_bounds__(256)
penh_build(const float* __restrict__ te, const float* __restrict__ pemb,
           const int* __restrict__ prio, half_t* __restrict__ out) {
  const int row = blockIdx.x;
  const int d   = threadIdx.x;
  int lvl = prio[row];
  out[(size_t)row * D_ + d] =
      (half_t)(te[(size_t)row * D_ + d] + pemb[(size_t)lvl * D_ + d]);
}

// sys16[b,t,:] = f16(se[b,:])
__global__ void __launch_bounds__(256)
sys_expand(const float* __restrict__ se, half_t* __restrict__ out) {
  const int row = blockIdx.x;
  const int d   = threadIdx.x;
  const int b   = row >> 9;
  out[(size_t)row * D_ + d] = (half_t)se[(size_t)b * D_ + d];
}

// f32 -> f16 copy
__global__ void __launch_bounds__(256)
cvt_f32_f16(const float* __restrict__ src, half_t* __restrict__ dst, int n) {
  int i = blockIdx.x * 256 + threadIdx.x;
  if (i < n) dst[i] = (half_t)src[i];
}

// mean over T of fused16 -> f16 [B,D]
__global__ void __launch_bounds__(256)
mean_T(const half_t* __restrict__ fused, half_t* __restrict__ outm) {
  int idx = blockIdx.x * 256 + threadIdx.x;
  if (idx < B_ * D_) {
    int b = idx >> 8, d = idx & 255;
    float s = 0.f;
    for (int t = 0; t < T_; ++t)
      s += (float)fused[((size_t)b * T_ + t) * D_ + d];
    outm[idx] = (half_t)(s * (1.f / (float)T_));
  }
}

// Tiny-N head: out[m,n] = act(X16[m,:] . W[n,:] + b[n])
__global__ void __launch_bounds__(256)
dot_small(const half_t* __restrict__ X, int K, int ldx,
          const float* __restrict__ W, const float* __restrict__ bias,
          float* __restrict__ out, int ldo, int M, int N, int act) {
  int idx = blockIdx.x * 256 + threadIdx.x;
  if (idx >= M * N) return;
  int m = idx / N, n = idx % N;
  float acc = bias ? bias[n] : 0.f;
  for (int k = 0; k < K; ++k)
    acc += (float)X[(size_t)m * ldx + k] * W[(size_t)n * K + k];
  if (act == 2) acc = tanhf(acc);
  out[(size_t)m * ldo + n] = acc;
}

// ---------------------------------------------------------------------------
// Host-side orchestration
// ---------------------------------------------------------------------------
struct GemmArgs {
  const half_t* A; int lda; long long saB, saH;
  const half_t* Bm; int ldb; long long sbB, sbH;
  float* Cf; half_t* Ch; int ldc; long long scB, scH;
  int M, N, K, heads;
  const float* bias; float alpha;
};

static void gemm(hipStream_t s, int act, int bt, int of16, int batches,
                 const GemmArgs& g) {
  dim3 grid((g.M + BM - 1) / BM, (g.N + BN - 1) / BN, batches);
#define GEMM_CASE(a, b, o)                                                   \
  if (act == a && bt == b && of16 == o) {                                    \
    gemm_wmma<a, b, o><<<grid, 256, 0, s>>>(                                 \
        g.A, g.lda, g.saB, g.saH, g.Bm, g.ldb, g.sbB, g.sbH, g.Cf, g.Ch,     \
        g.ldc, g.scB, g.scH, g.M, g.N, g.K, g.heads, g.bias, g.alpha);       \
    return;                                                                  \
  }
  GEMM_CASE(0, 0, 0) GEMM_CASE(0, 0, 1) GEMM_CASE(0, 1, 0) GEMM_CASE(0, 1, 1)
  GEMM_CASE(1, 0, 0) GEMM_CASE(1, 0, 1) GEMM_CASE(1, 1, 0) GEMM_CASE(1, 1, 1)
  GEMM_CASE(2, 0, 0) GEMM_CASE(2, 0, 1) GEMM_CASE(2, 1, 0) GEMM_CASE(2, 1, 1)
#undef GEMM_CASE
}

struct MhaScratch { half_t *Q, *K, *V, *O, *P; float* S; };

static void run_mha(hipStream_t s, const half_t* qin, const half_t* kvin,
                    int heads, int Tq, int Tk,
                    const half_t* qw, const float* qb,
                    const half_t* kw, const float* kb,
                    const half_t* vw, const float* vb,
                    const half_t* ow, const float* ob,
                    float* outF, half_t* outH, int ldc_out,
                    const MhaScratch& sc) {
  const int dk = D_ / heads;
  GemmArgs a;
  a = {qin, D_, 0, 0, qw, D_, 0, 0, nullptr, sc.Q, D_, 0, 0,
       B_ * Tq, D_, D_, 1, qb, 1.f};
  gemm(s, 0, 1, 1, 1, a);
  a = {kvin, D_, 0, 0, kw, D_, 0, 0, nullptr, sc.K, D_, 0, 0,
       B_ * Tk, D_, D_, 1, kb, 1.f};
  gemm(s, 0, 1, 1, 1, a);
  a = {kvin, D_, 0, 0, vw, D_, 0, 0, nullptr, sc.V, D_, 0, 0,
       B_ * Tk, D_, D_, 1, vb, 1.f};
  gemm(s, 0, 1, 1, 1, a);
  // scores = Q @ K^T / sqrt(dk)
  a = {sc.Q, D_, (long long)Tq * D_, dk,
       sc.K, D_, (long long)Tk * D_, dk,
       sc.S, nullptr, Tk, (long long)heads * Tq * Tk, (long long)Tq * Tk,
       Tq, Tk, dk, heads, nullptr, 1.f / sqrtf((float)dk)};
  gemm(s, 0, 1, 0, B_ * heads, a);
  softmax_rows<<<B_ * heads * Tq, 256, 0, s>>>(sc.S, sc.P, Tk);
  // O = P @ V (heads interleaved into [B,Tq,D])
  a = {sc.P, Tk, (long long)heads * Tq * Tk, (long long)Tq * Tk,
       sc.V, D_, (long long)Tk * D_, dk,
       nullptr, sc.O, D_, (long long)Tq * D_, dk,
       Tq, dk, Tk, heads, nullptr, 1.f};
  gemm(s, 0, 0, 1, B_ * heads, a);
  // output projection
  a = {sc.O, D_, 0, 0, ow, D_, 0, 0, outF, outH, ldc_out, 0, 0,
       B_ * Tq, D_, D_, 1, ob, 1.f};
  gemm(s, 0, 1, outH != nullptr, 1, a);
}

// params flattened indices (insertion order)
enum PIDX {
  I_TEW = 5, I_TEB, I_TEG, I_TEBE,
  I_REW, I_REB, I_REG, I_REBE,
  I_SEW, I_SEB, I_SEG, I_SEBE,
  I_TENC,
  I_DL_QW, I_DL_QB, I_DL_KW, I_DL_KB, I_DL_VW, I_DL_VB, I_DL_OW, I_DL_OB,
  I_URGW, I_URGB, I_PEMB,
  I_LV_QW, I_LV_QB, I_LV_KW, I_LV_KB, I_LV_VW, I_LV_VB, I_LV_OW, I_LV_OB,
  I_CR_QW, I_CR_QB, I_CR_KW, I_CR_KB, I_CR_VW, I_CR_VB, I_CR_OW, I_CR_OB,
  I_MA_QW, I_MA_QB, I_MA_KW, I_MA_KB, I_MA_VW, I_MA_VB, I_MA_OW, I_MA_OB,
  I_SA_QW, I_SA_QB, I_SA_KW, I_SA_KB, I_SA_VW, I_SA_VB, I_SA_OW, I_SA_OB,
  I_C1W, I_C1B, I_C2W, I_C2B, I_C3W, I_C3B,
  I_F1W, I_F1B, I_F2W, I_F2B,
  I_PA1W, I_PA1B, I_PA2W, I_PA2B,
  I_AS1W, I_AS1B, I_AS2W, I_AS2B,
  I_V1W, I_V1B, I_V2W, I_V2B
};

extern "C" void kernel_launch(void* const* d_in, const int* in_sizes, int n_in,
                              void* d_out, int out_size, void* d_ws,
                              size_t ws_size, hipStream_t stream) {
  const float* task = (const float*)d_in[0];
  const float* res  = (const float*)d_in[1];
  const float* sysf = (const float*)d_in[2];
  const float* dl   = (const float*)d_in[3];
  const int*   prio = (const int*)d_in[4];
  auto PF = [&](int i) { return (const float*)d_in[i]; };

  // ---- workspace bump allocator -------------------------------------------
  char* wsb = (char*)d_ws;
  size_t off = 0;
  auto allocB = [&](size_t bytes) -> void* {
    void* p = wsb + off;
    off = (off + bytes + 255) & ~(size_t)255;
    return p;
  };
  auto aF = [&](size_t n) { return (float*)allocB(n * 4); };
  auto aH = [&](size_t n) { return (half_t*)allocB(n * 2); };

  const size_t NT = (size_t)B_ * T_;  // 2048
  const size_t NR = (size_t)B_ * R_;  // 512

  float*  teF  = aF(NT * D_);   half_t* te16 = aH(NT * D_);
  float*  reF  = aF(NR * D_);   half_t* re16 = aH(NR * D_);
  float*  seF  = aF(B_ * D_);
  half_t* x16    = aH(NT * D_);
  half_t* penh16 = aH(NT * D_);
  half_t* sys16  = aH(NT * D_);
  half_t* lvl16  = aH(NT * D_);
  half_t* mo16   = aH(NT * D_);
  half_t* cat16  = aH(NT * 4 * D_);
  half_t* Q16 = aH(NT * D_); half_t* K16 = aH(NT * D_); half_t* V16 = aH(NT * D_);
  half_t* O16 = aH(NT * D_);
  float*  Sbuf = aF((size_t)B_ * 8 * T_ * T_);
  half_t* Pbuf = aH((size_t)B_ * 8 * T_ * T_);
  float*  aC = aF(NT * D_);
  float*  bC = aF(NR * D_);
  half_t* fused16 = aH(NT * D_);
  half_t* hf116   = aH(NT * D_);
  half_t* hpa16   = aH(NT * (D_ / 2));
  half_t* has16   = aH(NT * (D_ / 2));
  half_t* fmean16 = aH(B_ * D_);
  half_t* hv16    = aH(B_ * (D_ / 2));

  half_t* w16[85] = {};
  const int wlist[] = {I_DL_QW, I_DL_KW, I_DL_VW, I_DL_OW,
                       I_LV_QW, I_LV_KW, I_LV_VW, I_LV_OW,
                       I_CR_QW, I_CR_KW, I_CR_VW, I_CR_OW,
                       I_MA_QW, I_MA_KW, I_MA_VW, I_MA_OW,
                       I_SA_QW, I_SA_KW, I_SA_VW, I_SA_OW,
                       I_URGW, I_C1W, I_C2W, I_F1W, I_F2W,
                       I_PA1W, I_AS1W, I_AS2W, I_V1W};
  for (int i = 0; i < (int)(sizeof(wlist) / sizeof(int)); ++i)
    w16[wlist[i]] = aH((size_t)in_sizes[wlist[i]]);

  if (off > ws_size) return;  // workspace too small; bail deterministically

  for (int i = 0; i < (int)(sizeof(wlist) / sizeof(int)); ++i) {
    int n = in_sizes[wlist[i]];
    cvt_f32_f16<<<(n + 255) / 256, 256, 0, stream>>>(PF(wlist[i]),
                                                     w16[wlist[i]], n);
  }

  // ---- embeddings ----------------------------------------------------------
  embed_ln<<<(int)NT, 256, 0, stream>>>(task, 16, PF(I_TEW), PF(I_TEB),
                                        PF(I_TEG), PF(I_TEBE), teF, te16);
  embed_ln<<<(int)NR, 256, 0, stream>>>(res, 12, PF(I_REW), PF(I_REB),
                                        PF(I_REG), PF(I_REBE), reF, re16);
  half_t* se16tmp = fmean16;  // throwaway f16 target
  embed_ln<<<B_, 256, 0, stream>>>(sysf, 8, PF(I_SEW), PF(I_SEB), PF(I_SEG),
                                   PF(I_SEBE), seF, se16tmp);

  MhaScratch sc{Q16, K16, V16, O16, Pbuf, Sbuf};
  GemmArgs g;

  // ---- temporal stream -> cat16 cols [0,256) ------------------------------
  temporal_x<<<(int)NT, 256, 0, stream>>>(teF, PF(I_TENC), dl, x16);
  run_mha(stream, x16, x16, 8, T_, T_,
          w16[I_DL_QW], PF(I_DL_QB), w16[I_DL_KW], PF(I_DL_KB),
          w16[I_DL_VW], PF(I_DL_VB), w16[I_DL_OW], PF(I_DL_OB),
          nullptr, mo16, D_, sc);
  g = {mo16, D_, 0, 0, w16[I_URGW], D_, 0, 0, nullptr, cat16 + 0, 4 * D_, 0, 0,
       (int)NT, D_, D_, 1, PF(I_URGB), 1.f};
  gemm(stream, 0, 1, 1, 1, g);

  // ---- hierarchical stream -> cat16 cols [256,512) ------------------------
  penh_build<<<(int)NT, 256, 0, stream>>>(teF, PF(I_PEMB), prio, penh16);
  run_mha(stream, penh16, penh16, 4, T_, T_,
          w16[I_LV_QW], PF(I_LV_QB), w16[I_LV_KW], PF(I_LV_KB),
          w16[I_LV_VW], PF(I_LV_VB), w16[I_LV_OW], PF(I_LV_OB),
          nullptr, lvl16, D_, sc);
  sys_expand<<<(int)NT, 256, 0, stream>>>(seF, sys16);
  run_mha(stream, lvl16, sys16, 8, T_, T_,
          w16[I_CR_QW], PF(I_CR_QB), w16[I_CR_KW], PF(I_CR_KB),
          w16[I_CR_VW], PF(I_CR_VB), w16[I_CR_OW], PF(I_CR_OB),
          nullptr, cat16 + D_, 4 * D_, sc);

  // ---- task-resource stream -> cat16 cols [512,768) -----------------------
  run_mha(stream, te16, re16, 8, T_, R_,
          w16[I_MA_QW], PF(I_MA_QB), w16[I_MA_KW], PF(I_MA_KB),
          w16[I_MA_VW], PF(I_MA_VB), w16[I_MA_OW], PF(I_MA_OB),
          nullptr, cat16 + 2 * D_, 4 * D_, sc);

  // ---- self stream -> cat16 cols [768,1024) -------------------------------
  run_mha(stream, te16, te16, 8, T_, T_,
          w16[I_SA_QW], PF(I_SA_QB), w16[I_SA_KW], PF(I_SA_KB),
          w16[I_SA_VW], PF(I_SA_VB), w16[I_SA_OW], PF(I_SA_OB),
          nullptr, cat16 + 3 * D_, 4 * D_, sc);

  // ---- compat: aC = te @ c1[:, :D]^T, bC = re @ c1[:, D:]^T ---------------
  g = {te16, D_, 0, 0, w16[I_C1W], 2 * D_, 0, 0, aC, nullptr, D_, 0, 0,
       (int)NT, D_, D_, 1, nullptr, 1.f};
  gemm(stream, 0, 1, 0, 1, g);
  g = {re16, D_, 0, 0, w16[I_C1W] + D_, 2 * D_, 0, 0, bC, nullptr, D_, 0, 0,
       (int)NR, D_, D_, 1, nullptr, 1.f};
  gemm(stream, 0, 1, 0, 1, g);

  float* out_assign = (float*)d_out;
  float* out_priadj = out_assign + (size_t)B_ * T_ * R_;
  float* out_vals   = out_priadj + (size_t)B_ * T_;
  float* out_compat = out_vals + (size_t)B_ * 3;

  compat_fused<<<dim3(T_ / 16, R_ / 4, B_), 256, 0, stream>>>(
      aC, bC, PF(I_C1B), w16[I_C2W], PF(I_C2B), PF(I_C3W), PF(I_C3B),
      out_compat);

  // ---- fusion -------------------------------------------------------------
  g = {cat16, 4 * D_, 0, 0, w16[I_F1W], 4 * D_, 0, 0, nullptr, hf116, D_, 0, 0,
       (int)NT, D_, 4 * D_, 1, PF(I_F1B), 1.f};
  gemm(stream, 1, 1, 1, 1, g);  // relu
  g = {hf116, D_, 0, 0, w16[I_F2W], D_, 0, 0, nullptr, fused16, D_, 0, 0,
       (int)NT, D_, D_, 1, PF(I_F2B), 1.f};
  gemm(stream, 2, 1, 1, 1, g);  // tanh

  // ---- heads --------------------------------------------------------------
  g = {fused16, D_, 0, 0, w16[I_PA1W], D_, 0, 0, nullptr, hpa16, D_ / 2, 0, 0,
       (int)NT, D_ / 2, D_, 1, PF(I_PA1B), 1.f};
  gemm(stream, 1, 1, 1, 1, g);
  dot_small<<<((int)NT + 255) / 256, 256, 0, stream>>>(
      hpa16, D_ / 2, D_ / 2, PF(I_PA2W), PF(I_PA2B), out_priadj, 1, (int)NT, 1,
      2 /*tanh*/);

  g = {fused16, D_, 0, 0, w16[I_AS1W], D_, 0, 0, nullptr, has16, D_ / 2, 0, 0,
       (int)NT, D_ / 2, D_, 1, PF(I_AS1B), 1.f};
  gemm(stream, 1, 1, 1, 1, g);
  g = {has16, D_ / 2, 0, 0, w16[I_AS2W], D_ / 2, 0, 0, out_assign, nullptr, R_,
       0, 0, (int)NT, R_, D_ / 2, 1, PF(I_AS2B), 1.f};
  gemm(stream, 0, 1, 0, 1, g);

  mean_T<<<(B_ * D_ + 255) / 256, 256, 0, stream>>>(fused16, fmean16);
  g = {fmean16, D_, 0, 0, w16[I_V1W], D_, 0, 0, nullptr, hv16, D_ / 2, 0, 0,
       B_, D_ / 2, D_, 1, PF(I_V1B), 1.f};
  gemm(stream, 1, 1, 1, 1, g);
  dot_small<<<1, 256, 0, stream>>>(hv16, D_ / 2, D_ / 2, PF(I_V2W), PF(I_V2B),
                                   out_vals, 3, B_, 3, 0);
}
